// MultiVariateHawkesProcessModel_37125697307223
// MI455X (gfx1250) — matrix operand order repaired and checked by reading, not compiled, not run on hardware
//
#include <hip/hip_runtime.h>
#include <hip/hip_bf16.h>

typedef __attribute__((ext_vector_type(16))) _Float16 v16h;
typedef __attribute__((ext_vector_type(8)))  float    v8f;

#define B_   4
#define N_   2048
#define NT_  50
#define ND_  4
#define K_   200          // NT_*ND_
#define KP_  224          // K padded to 7 chunks of 32
#define NC_  64           // output cols padded to 4 tiles of 16
#define M_   (B_*N_)      // 8192 rows
#define KCH_ (KP_/32)     // 7 WMMA K-chunks

__device__ __forceinline__ float softplus_f(float x) {
    // numerically stable: max(x,0) + log1p(exp(-|x|))
    return fmaxf(x, 0.0f) + log1pf(__expf(-fabsf(x)));
}

// ---------------------------------------------------------------------------
// Phase 1: parameter transforms.
//   mhat = softplus(mu_p)            (50)
//   omega = softplus(beta_p)         (4)
//   Wt[m][j] = softplus(alpha_p[m*200+j])  transposed weight (64 x 224, f16)
//   colsum[j] = sum_m softplus(alpha_p[m*200+j])   (200)
//   summhat  = sum(mhat)
// ---------------------------------------------------------------------------
__global__ void hawkes_prep(const float* __restrict__ mu_p,
                            const float* __restrict__ alpha_p,
                            const float* __restrict__ beta_p,
                            float* __restrict__ mhat,
                            float* __restrict__ omega,
                            float* __restrict__ colsum,
                            float* __restrict__ summhat,
                            _Float16* __restrict__ Wt) {
    __shared__ float smh[64];
    int tid = threadIdx.x;
    if (tid < 64) smh[tid] = 0.0f;
    __syncthreads();
    if (tid < NT_) {
        float v = softplus_f(mu_p[tid]);
        mhat[tid] = v;
        smh[tid] = v;
    }
    if (tid < ND_) omega[tid] = softplus_f(beta_p[tid]);
    __syncthreads();
    if (tid == 0) {
        float s = 0.0f;
        for (int i = 0; i < NT_; ++i) s += smh[i];
        *summhat = s;
    }
    for (int idx = tid; idx < NC_ * KP_; idx += (int)blockDim.x) {
        int m = idx / KP_;
        int j = idx - m * KP_;
        float v = (m < NT_ && j < K_) ? softplus_f(alpha_p[m * K_ + j]) : 0.0f;
        Wt[idx] = (_Float16)v;
    }
    if (tid < K_) {
        float s = 0.0f;
        for (int m = 0; m < NT_; ++m) s += softplus_f(alpha_p[m * K_ + tid]);
        colsum[tid] = s;
    }
}

// ---------------------------------------------------------------------------
// Phase 2: Markovian scan. One block per batch, 224 threads.
// Thread tid<200 owns (k = tid/4, d = tid%4) and maintains
//   R(i) = sum_{j<i, ty_j=k} exp(-omega_d (t_i - t_j))
// writing G[row][tid] = omega_d * R(i) as f16 (WMMA A operand).
// Threads 200..223 write the zero K-padding.
// ---------------------------------------------------------------------------
__global__ void hawkes_scan(const float* __restrict__ times,
                            const int*   __restrict__ types,
                            const float* __restrict__ omega,
                            _Float16*    __restrict__ G) {
    __shared__ float st[N_];
    __shared__ int   sy[N_];
    const int b   = blockIdx.x;
    const int tid = threadIdx.x;
    for (int i = tid; i < N_; i += (int)blockDim.x) {
        st[i] = times[b * N_ + i];
        sy[i] = types[b * N_ + i];
    }
    __syncthreads();
    _Float16* Gb = G + (size_t)b * N_ * KP_;
    if (tid < K_) {
        const int   kk = tid >> 2;
        const float om = omega[tid & 3];
        float R = 0.0f;
        float tprev = st[0];
        for (int i = 0; i < N_; ++i) {
            float t = st[i];
            if (i > 0) {
                float add = (sy[i - 1] == kk) ? 1.0f : 0.0f;
                R = (R + add) * __expf(-om * (t - tprev));
            }
            Gb[(size_t)i * KP_ + tid] = (_Float16)(om * R);
            tprev = t;
        }
    } else {
        for (int i = 0; i < N_; ++i)
            Gb[(size_t)i * KP_ + tid] = (_Float16)0.0f;
    }
}

// ---------------------------------------------------------------------------
// Phase 3: P = G (8192 x 224, f16) @ Wt^T (224 x 64, f16) -> f32.
// 64 blocks x 8 waves; each wave owns one M-tile of 16 rows and computes
// ALL 4 N-tiles, reusing its A fragment in registers across 4 WMMAs per
// K-chunk (4 independent accumulator chains hide XDL latency; A traffic
// cut 4x vs one-tile-per-wave). A and B fragments use contiguous k = 16*sel + s.
// ---------------------------------------------------------------------------
__global__ void __launch_bounds__(256)
hawkes_gemm(const _Float16* __restrict__ G,
            const _Float16* __restrict__ Wt,
            float* __restrict__ P) {
    const int wave  = threadIdx.x >> 5;              // 0..7
    const int lane  = threadIdx.x & 31;
    const int mtile = blockIdx.x * 8 + wave;         // 0..511
    const int l15   = lane & 15;
    const int sel   = lane >> 4;                     // lane-group

    // A: lane l15 = row M; lane-group selects K base 0/16 (contiguous 16 K)
    const _Float16* Arow = G + (size_t)(mtile * 16 + l15) * KP_;
    // B: lane l15 = col N; lane-group selects K base 0/16 (Wt stored N-major)
    const _Float16* Bbase = Wt + (size_t)l15 * KP_ + sel * 16;

    v8f acc0 = {0.f,0.f,0.f,0.f,0.f,0.f,0.f,0.f};
    v8f acc1 = acc0, acc2 = acc0, acc3 = acc0;

    for (int kc = 0; kc < KCH_; ++kc) {
        const int kb = kc * 32;
        if (kc + 1 < KCH_) {
            __builtin_prefetch(Arow + kb + 32, 0, 1);  // global_prefetch_b8
        }
        v16h a;
#pragma unroll
        for (int h = 0; h < 8; ++h) a[h]     = Arow[kb + sel * 16 + h];
#pragma unroll
        for (int h = 0; h < 8; ++h) a[8 + h] = Arow[kb + sel * 16 + 8 + h];

        v16h b0, b1, b2, b3;
#pragma unroll
        for (int h = 0; h < 16; ++h) b0[h] = Bbase[(size_t)(0 * 16) * KP_ + kb + h];
#pragma unroll
        for (int h = 0; h < 16; ++h) b1[h] = Bbase[(size_t)(1 * 16) * KP_ + kb + h];
#pragma unroll
        for (int h = 0; h < 16; ++h) b2[h] = Bbase[(size_t)(2 * 16) * KP_ + kb + h];
#pragma unroll
        for (int h = 0; h < 16; ++h) b3[h] = Bbase[(size_t)(3 * 16) * KP_ + kb + h];

        acc0 = __builtin_amdgcn_wmma_f32_16x16x32_f16(false, a, false, b0,
                                                      (short)0, acc0, false, false);
        acc1 = __builtin_amdgcn_wmma_f32_16x16x32_f16(false, a, false, b1,
                                                      (short)0, acc1, false, false);
        acc2 = __builtin_amdgcn_wmma_f32_16x16x32_f16(false, a, false, b2,
                                                      (short)0, acc2, false, false);
        acc3 = __builtin_amdgcn_wmma_f32_16x16x32_f16(false, a, false, b3,
                                                      (short)0, acc3, false, false);
    }

    // D layout: lane l15 = col N; VGPR r -> row M = sel*8 + r
    float* Pt = P + (size_t)(mtile * 16 + sel * 8) * NC_ + l15;
#pragma unroll
    for (int r = 0; r < 8; ++r) {
        float* row = Pt + (size_t)r * NC_;
        row[0 * 16] = acc0[r];
        row[1 * 16] = acc1[r];
        row[2 * 16] = acc2[r];
        row[3 * 16] = acc3[r];
    }
}

// ---------------------------------------------------------------------------
// Phase 4: gather + reductions. One block per batch.
//   loglik[b] = sum_i log(mhat[ty_i] + P[row_i, ty_i] + 1e-8) * mask_i
//   comp[b]   = (t1-t0)*summhat + sum_j mask_j * sum_d colsum[ty_j,d]*(1-e^{-w_d(t1-t_j)})
// ---------------------------------------------------------------------------
__global__ void hawkes_final(const float* __restrict__ times,
                             const int*   __restrict__ types,
                             const float* __restrict__ mask,
                             const float* __restrict__ t0,
                             const float* __restrict__ t1,
                             const float* __restrict__ P,
                             const float* __restrict__ mhat,
                             const float* __restrict__ omega,
                             const float* __restrict__ colsum,
                             const float* __restrict__ summhat,
                             float* __restrict__ out) {
    __shared__ float r_ll[256];
    __shared__ float r_cp[256];
    const int b = blockIdx.x, tid = threadIdx.x;
    const float T1 = t1[b], T0 = t0[b];
    const float om0 = omega[0], om1 = omega[1], om2 = omega[2], om3 = omega[3];
    float all = 0.0f, acp = 0.0f;
    for (int i = tid; i < N_; i += 256) {
        const int idx = b * N_ + i;
        const int ty  = types[idx];
        const float mk = mask[idx];
        const float rate = mhat[ty] + P[(size_t)idx * NC_ + ty];
        all += logf(rate + 1e-8f) * mk;
        const float dtj = T1 - times[idx];
        float s = colsum[ty * 4 + 0] * (1.0f - __expf(-om0 * dtj))
                + colsum[ty * 4 + 1] * (1.0f - __expf(-om1 * dtj))
                + colsum[ty * 4 + 2] * (1.0f - __expf(-om2 * dtj))
                + colsum[ty * 4 + 3] * (1.0f - __expf(-om3 * dtj));
        acp += mk * s;
    }
    r_ll[tid] = all;
    r_cp[tid] = acp;
    __syncthreads();
    for (int s = 128; s > 0; s >>= 1) {
        if (tid < s) { r_ll[tid] += r_ll[tid + s]; r_cp[tid] += r_cp[tid + s]; }
        __syncthreads();
    }
    if (tid == 0) {
        out[b]      = r_ll[0];
        out[B_ + b] = (T1 - T0) * (*summhat) + r_cp[0];
    }
}

extern "C" void kernel_launch(void* const* d_in, const int* in_sizes, int n_in,
                              void* d_out, int out_size, void* d_ws, size_t ws_size,
                              hipStream_t stream) {
    const float* times   = (const float*)d_in[0];
    const int*   types   = (const int*)  d_in[1];
    const float* mask    = (const float*)d_in[2];
    const float* t0      = (const float*)d_in[3];
    const float* t1      = (const float*)d_in[4];
    const float* mu_p    = (const float*)d_in[5];
    const float* alpha_p = (const float*)d_in[6];
    const float* beta_p  = (const float*)d_in[7];
    float* out = (float*)d_out;

    char* ws = (char*)d_ws;
    size_t offG  = 0;
    size_t offWt = offG  + (size_t)M_ * KP_ * sizeof(_Float16);   // 3.50 MB
    size_t offP  = offWt + (size_t)NC_ * KP_ * sizeof(_Float16);  // +28 KB
    size_t offMh = offP  + (size_t)M_ * NC_ * sizeof(float);      // +2 MB
    size_t offOm = offMh + 64  * sizeof(float);
    size_t offCs = offOm + 16  * sizeof(float);
    size_t offSm = offCs + 256 * sizeof(float);
    (void)offSm; (void)ws_size; (void)in_sizes; (void)n_in; (void)out_size;

    _Float16* G      = (_Float16*)(ws + offG);
    _Float16* Wt     = (_Float16*)(ws + offWt);
    float*    P      = (float*)   (ws + offP);
    float*    mhat   = (float*)   (ws + offMh);
    float*    omega  = (float*)   (ws + offOm);
    float*    colsum = (float*)   (ws + offCs);
    float*    summ   = (float*)   (ws + offSm);

    hipLaunchKernelGGL(hawkes_prep,  dim3(1),       dim3(256), 0, stream,
                       mu_p, alpha_p, beta_p, mhat, omega, colsum, summ, Wt);
    hipLaunchKernelGGL(hawkes_scan,  dim3(B_),      dim3(KP_), 0, stream,
                       times, types, omega, G);
    hipLaunchKernelGGL(hawkes_gemm,  dim3(M_ / (16 * 8)), dim3(256), 0, stream,
                       G, Wt, P);
    hipLaunchKernelGGL(hawkes_final, dim3(B_),      dim3(256), 0, stream,
                       times, types, mask, t0, t1, P, mhat, omega, colsum, summ, out);
}
